// CategoricalEmbedding_41721312314019
// MI455X (gfx1250) — compile-verified, matrix operand order checked by reading, not run on hardware
//
#include <hip/hip_runtime.h>

#define N_ROWS 16384
#define DIMS   512
#define KCODES 4096

typedef __bf16 bf16_t;
typedef __attribute__((ext_vector_type(16))) __bf16 v16bf;
typedef __attribute__((ext_vector_type(8)))  __bf16 v8bf;
typedef __attribute__((ext_vector_type(4)))  __bf16 v4bf;
typedef __attribute__((ext_vector_type(8)))  float  v8f;
typedef __attribute__((ext_vector_type(4)))  int    v4i;

union V16 { v16bf v; v8bf h[2]; };

// ---- CDNA5 async global->LDS path (guarded; falls back to plain copies) ----
#if __has_builtin(__builtin_amdgcn_global_load_async_to_lds_b128)
  #define HAVE_ASYNC_LDS 1
#else
  #define HAVE_ASYNC_LDS 0
#endif

#if __has_builtin(__builtin_amdgcn_s_wait_asynccnt)
  #define WAIT_ASYNC(n) __builtin_amdgcn_s_wait_asynccnt(n)
#elif HAVE_ASYNC_LDS
  #define WAIT_ASYNC(n) asm volatile("s_wait_asynccnt %0" :: "n"(n) : "memory")
#else
  #define WAIT_ASYNC(n) ((void)0)
#endif

__device__ __forceinline__ void async_cp16(const bf16_t* g, bf16_t* l){
#if HAVE_ASYNC_LDS
  // param types (from hipcc diagnostic): int4 AS1 * source, int4 AS3 * dest
  __builtin_amdgcn_global_load_async_to_lds_b128(
      (__attribute__((address_space(1))) v4i*)(g),
      (__attribute__((address_space(3))) v4i*)(l), 0, 0);
#else
  *(v8bf*)l = *(const v8bf*)g;     // plain 16B copy through VGPRs
#endif
}

__device__ __forceinline__ unsigned f2u(float f){ union{float f;unsigned u;}c;c.f=f;return c.u;}
__device__ __forceinline__ float u2f(unsigned u){ union{float f;unsigned u;}c;c.u=u;return c.f;}
__device__ __forceinline__ bf16_t bfbits(unsigned short s){ union{unsigned short s;bf16_t b;}c;c.s=s;return c.b;}

// round-to-nearest-even split: f ~= hi + lo, both bf16 (~16 mantissa bits total)
__device__ __forceinline__ void split2(float f, bf16_t* hi, bf16_t* lo){
  unsigned u = f2u(f);
  unsigned r = u + 0x7FFFu + ((u >> 16) & 1u);
  unsigned short hs = (unsigned short)(r >> 16);
  *hi = bfbits(hs);
  float rem = f - u2f(((unsigned)hs) << 16);
  unsigned ur = f2u(rem);
  unsigned rr = ur + 0x7FFFu + ((ur >> 16) & 1u);
  *lo = bfbits((unsigned short)(rr >> 16));
}

// ---------------- shared WMMA core ----------------
// One D-step: load A/B frags (padded stride-40 rows) and run 12 bf16 WMMAs
// (hi*hi + hi*lo + lo*hi) on a 2x2 grid of 16x16 f32 accumulators.
__device__ __forceinline__ void mma_step(const bf16_t* aH, const bf16_t* aL,
                                         const bf16_t* bH, const bf16_t* bL,
                                         int wm, int wn, int half, int col,
                                         v8f (&acc)[2][2])
{
  V16 ah[2], al[2], bh[2], bl[2];
  #pragma unroll
  for (int m = 0; m < 2; ++m){
    int arow = wm*32 + m*16 + col;
    ah[m].h[0] = *(const v8bf*)&aH[arow*40 + half*8];
    ah[m].h[1] = *(const v8bf*)&aH[arow*40 + 16 + half*8];
    al[m].h[0] = *(const v8bf*)&aL[arow*40 + half*8];
    al[m].h[1] = *(const v8bf*)&aL[arow*40 + 16 + half*8];
  }
  #pragma unroll
  for (int n = 0; n < 2; ++n){
    int brow = wn*32 + n*16 + col;
    bh[n].h[0] = *(const v8bf*)&bH[brow*40 + half*8];
    bh[n].h[1] = *(const v8bf*)&bH[brow*40 + 16 + half*8];
    bl[n].h[0] = *(const v8bf*)&bL[brow*40 + half*8];
    bl[n].h[1] = *(const v8bf*)&bL[brow*40 + 16 + half*8];
  }
  #pragma unroll
  for (int m = 0; m < 2; ++m)
    #pragma unroll
    for (int n = 0; n < 2; ++n){
      acc[m][n] = __builtin_amdgcn_wmma_f32_16x16x32_bf16(
          false, ah[m].v, false, bh[n].v, (short)0, acc[m][n], false, false);
      acc[m][n] = __builtin_amdgcn_wmma_f32_16x16x32_bf16(
          false, ah[m].v, false, bl[n].v, (short)0, acc[m][n], false, false);
      acc[m][n] = __builtin_amdgcn_wmma_f32_16x16x32_bf16(
          false, al[m].v, false, bh[n].v, (short)0, acc[m][n], false, false);
    }
}

// dist = ||c||^2 - 2 x.c  folded into running per-(row,lane-col) argmin.
// ascending code order + strict '<' keeps the first (lowest) index.
__device__ __forceinline__ void score_update(const float* __restrict__ cnorm,
                                             int kc, int wn, int col,
                                             v8f (&acc)[2][2],
                                             float (&minv)[2][8], int (&mini)[2][8])
{
  #pragma unroll
  for (int n = 0; n < 2; ++n){
    int code = kc + wn*32 + n*16 + col;
    float cn = cnorm[code];
    #pragma unroll
    for (int m = 0; m < 2; ++m)
      #pragma unroll
      for (int r = 0; r < 8; ++r){
        float d = cn - 2.0f * acc[m][n][r];
        if (d < minv[m][r]) { minv[m][r] = d; mini[m][r] = code; }
      }
  }
}

__device__ __forceinline__ void argmin_finalize(float (&minv)[2][8], int (&mini)[2][8],
                                                int wm, int wn, int half, int col, int tid,
                                                float (*redv)[64], int (*redi)[64],
                                                float* __restrict__ out_idx, int row0)
{
  // butterfly over the 16 column-lanes of each half (xor masks stay in-half)
  #pragma unroll
  for (int mask = 1; mask < 16; mask <<= 1){
    #pragma unroll
    for (int m = 0; m < 2; ++m)
      #pragma unroll
      for (int r = 0; r < 8; ++r){
        float ov = __shfl_xor(minv[m][r], mask, 32);
        int   oi = __shfl_xor(mini[m][r], mask, 32);
        if (ov < minv[m][r] || (ov == minv[m][r] && oi < mini[m][r])){
          minv[m][r] = ov; mini[m][r] = oi;
        }
      }
  }
  if (col == 0){
    #pragma unroll
    for (int m = 0; m < 2; ++m)
      #pragma unroll
      for (int r = 0; r < 8; ++r){
        int row = wm*32 + m*16 + half*8 + r;
        redv[wn][row] = minv[m][r];
        redi[wn][row] = mini[m][r];
      }
  }
  __syncthreads();
  if (tid < 64){
    float bv = redv[0][tid]; int bi = redi[0][tid];
    #pragma unroll
    for (int w = 1; w < 4; ++w){
      float v = redv[w][tid]; int i = redi[w][tid];
      if (v < bv || (v == bv && i < bi)){ bv = v; bi = i; }
    }
    out_idx[(size_t)row0 + tid] = (float)bi;
  }
}

// ---------------- Kernel: codebook squared norms ----------------
__global__ __launch_bounds__(256) void vq_cnorm_kernel(const float* __restrict__ cb,
                                                       float* __restrict__ cnorm){
  int k = blockIdx.x * blockDim.x + threadIdx.x;
  if (k >= KCODES) return;
  const float4* row = (const float4*)(cb + (size_t)k * DIMS);
  float s = 0.f;
  #pragma unroll 8
  for (int i = 0; i < DIMS/4; ++i){
    float4 v = row[i];
    s += v.x*v.x + v.y*v.y + v.z*v.z + v.w*v.w;
  }
  cnorm[k] = s;
}

// ---------------- Kernel: fp32 -> bf16 hi/lo planes (prep for fast path) ----
__global__ __launch_bounds__(256) void vq_split_kernel(const float* __restrict__ src,
                                                       bf16_t* __restrict__ hi,
                                                       bf16_t* __restrict__ lo, int n4){
  int i = blockIdx.x * blockDim.x + threadIdx.x;
  if (i >= n4) return;
  float4 v = ((const float4*)src)[i];
  float t[4] = {v.x, v.y, v.z, v.w};
  v4bf h, l;
  #pragma unroll
  for (int e = 0; e < 4; ++e){
    bf16_t hh, ll;
    split2(t[e], &hh, &ll);
    h[e] = hh; l[e] = ll;
  }
  *(v4bf*)(hi + (size_t)i*4) = h;
  *(v4bf*)(lo + (size_t)i*4) = l;
}

// ---------------- Fast argmin: pre-split bf16 + async LDS double-buffer ----
// 256 threads = 8 waves; block owns 64 rows; 512 pipeline steps
// (32 code panels x 16 D-steps). Stage s fills buf s&1 for step s.
__global__ __launch_bounds__(256) void vq_argmin_fast_kernel(
    const bf16_t* __restrict__ xh, const bf16_t* __restrict__ xl,
    const bf16_t* __restrict__ ch, const bf16_t* __restrict__ cl,
    const float* __restrict__ cnorm, float* __restrict__ out_idx)
{
  __shared__ __align__(16) bf16_t sAh[2][64 * 40];
  __shared__ __align__(16) bf16_t sAl[2][64 * 40];
  __shared__ __align__(16) bf16_t sBh[2][128 * 40];
  __shared__ __align__(16) bf16_t sBl[2][128 * 40];
  __shared__ float redv[4][64];
  __shared__ int   redi[4][64];

  const int tid  = threadIdx.x;
  const int lane = tid & 31;
  const int wave = tid >> 5;
  const int wm   = wave & 1;
  const int wn   = wave >> 1;
  const int half = (lane >> 4) & 1;
  const int col  = lane & 15;
  const int row0 = blockIdx.x * 64;

  const int ar = tid >> 2;        // 0..63 : A row
  const int aj = (tid & 3) * 8;   // 16B chunk within 32-dim slice
  const int bc = tid >> 1;        // 0..127: B code
  const int bj = (tid & 1) * 16;  // 32B (two 16B chunks)

  float minv[2][8]; int mini[2][8];
  #pragma unroll
  for (int m = 0; m < 2; ++m)
    #pragma unroll
    for (int r = 0; r < 8; ++r){ minv[m][r] = 3.402823466e38f; mini[m][r] = 0; }

  v8f acc[2][2];

  { // prologue: stage step 0 into buf 0 (6 x async b128 per thread)
    size_t aoff = (size_t)(row0 + ar) * DIMS + aj;
    async_cp16(xh + aoff, &sAh[0][ar*40 + aj]);
    async_cp16(xl + aoff, &sAl[0][ar*40 + aj]);
    size_t boff = (size_t)bc * DIMS + bj;
    async_cp16(ch + boff,     &sBh[0][bc*40 + bj]);
    async_cp16(ch + boff + 8, &sBh[0][bc*40 + bj + 8]);
    async_cp16(cl + boff,     &sBl[0][bc*40 + bj]);
    async_cp16(cl + boff + 8, &sBl[0][bc*40 + bj + 8]);
  }

  for (int s = 0; s < 512; ++s){
    const int buf = s & 1;
    const int kc  = (s >> 4) * 128;

    if ((s & 15) == 0){
      #pragma unroll
      for (int m = 0; m < 2; ++m)
        #pragma unroll
        for (int n = 0; n < 2; ++n){
          v8f z = {0.f,0.f,0.f,0.f,0.f,0.f,0.f,0.f};
          acc[m][n] = z;
        }
    }

    __syncthreads();                 // all waves done reading buf^1
    if (s + 1 < 512){                // prefetch next step into buf^1
      const int nb  = buf ^ 1;
      const int kc2 = ((s + 1) >> 4) * 128;
      const int ds2 = ((s + 1) & 15) * 32;
      size_t aoff = (size_t)(row0 + ar) * DIMS + ds2 + aj;
      async_cp16(xh + aoff, &sAh[nb][ar*40 + aj]);
      async_cp16(xl + aoff, &sAl[nb][ar*40 + aj]);
      size_t boff = (size_t)(kc2 + bc) * DIMS + ds2 + bj;
      async_cp16(ch + boff,     &sBh[nb][bc*40 + bj]);
      async_cp16(ch + boff + 8, &sBh[nb][bc*40 + bj + 8]);
      async_cp16(cl + boff,     &sBl[nb][bc*40 + bj]);
      async_cp16(cl + boff + 8, &sBl[nb][bc*40 + bj + 8]);
      WAIT_ASYNC(6);                 // current buf's 6 loads retired (in-order)
    } else {
      WAIT_ASYNC(0);
    }
    __syncthreads();                 // current buf visible to all waves

    mma_step(sAh[buf], sAl[buf], sBh[buf], sBl[buf], wm, wn, half, col, acc);

    if ((s & 15) == 15)
      score_update(cnorm, kc, wn, col, acc, minv, mini);
  }

  argmin_finalize(minv, mini, wm, wn, half, col, tid, redv, redi, out_idx, row0);
}

// ---------------- Fallback argmin: in-kernel conversion (small d_ws) ----------
__global__ __launch_bounds__(256) void vq_argmin_fallback_kernel(
    const float* __restrict__ x, const float* __restrict__ cb,
    const float* __restrict__ cnorm, float* __restrict__ out_idx)
{
  __shared__ __align__(16) bf16_t sAh[64 * 40];
  __shared__ __align__(16) bf16_t sAl[64 * 40];
  __shared__ __align__(16) bf16_t sBh[128 * 40];
  __shared__ __align__(16) bf16_t sBl[128 * 40];
  __shared__ float redv[4][64];
  __shared__ int   redi[4][64];

  const int tid  = threadIdx.x;
  const int lane = tid & 31;
  const int wave = tid >> 5;
  const int wm   = wave & 1;
  const int wn   = wave >> 1;
  const int half = (lane >> 4) & 1;
  const int col  = lane & 15;
  const int row0 = blockIdx.x * 64;

  const float* xb = x + (size_t)row0 * DIMS;

  float minv[2][8]; int mini[2][8];
  #pragma unroll
  for (int m = 0; m < 2; ++m)
    #pragma unroll
    for (int r = 0; r < 8; ++r){ minv[m][r] = 3.402823466e38f; mini[m][r] = 0; }

  const int ar = tid >> 2, aj = (tid & 3) * 8;
  const int bc = tid >> 1, bj = (tid & 1) * 16;

  for (int kc = 0; kc < KCODES; kc += 128) {
    v8f acc[2][2];
    #pragma unroll
    for (int m = 0; m < 2; ++m)
      #pragma unroll
      for (int n = 0; n < 2; ++n){
        v8f z = {0.f,0.f,0.f,0.f,0.f,0.f,0.f,0.f};
        acc[m][n] = z;
      }

    for (int ds = 0; ds < DIMS; ds += 32) {
      __syncthreads();
      {
        const float* src = xb + (size_t)ar * DIMS + ds + aj;
        float4 a0 = *(const float4*)(src);
        float4 a1 = *(const float4*)(src + 4);
        float t[8] = {a0.x,a0.y,a0.z,a0.w,a1.x,a1.y,a1.z,a1.w};
        bf16_t* dh = &sAh[ar*40 + aj];
        bf16_t* dl = &sAl[ar*40 + aj];
        #pragma unroll
        for (int q = 0; q < 8; ++q) split2(t[q], dh + q, dl + q);
      }
      {
        const float* src = cb + (size_t)(kc + bc) * DIMS + ds + bj;
        bf16_t* dh = &sBh[bc*40 + bj];
        bf16_t* dl = &sBl[bc*40 + bj];
        #pragma unroll
        for (int q = 0; q < 4; ++q){
          float4 v = *(const float4*)(src + q*4);
          float t[4] = {v.x,v.y,v.z,v.w};
          #pragma unroll
          for (int e = 0; e < 4; ++e) split2(t[e], dh + q*4 + e, dl + q*4 + e);
        }
      }
      __syncthreads();
      mma_step(sAh, sAl, sBh, sBl, wm, wn, half, col, acc);
    }
    score_update(cnorm, kc, wn, col, acc, minv, mini);
  }

  argmin_finalize(minv, mini, wm, wn, half, col, tid, redv, redi, out_idx, row0);
}

// ---------------- Kernel: gather codes -> z_st and z_st_bar ----------------
__global__ __launch_bounds__(128) void vq_gather_kernel(
    const float* __restrict__ cb, const float* __restrict__ idxf,
    float* __restrict__ z0, float* __restrict__ z1)
{
  int row = blockIdx.x;
  int k = (int)idxf[row];
  const float4* s = (const float4*)(cb + (size_t)k * DIMS);
  float4* d0 = (float4*)(z0 + (size_t)row * DIMS);
  float4* d1 = (float4*)(z1 + (size_t)row * DIMS);
  int i = threadIdx.x;              // 128 threads == DIMS/4 float4s
  float4 v = s[i];
  d0[i] = v;
  d1[i] = v;
}

extern "C" void kernel_launch(void* const* d_in, const int* in_sizes, int n_in,
                              void* d_out, int out_size, void* d_ws, size_t ws_size,
                              hipStream_t stream) {
  const float* x  = (const float*)d_in[0];   // [8,2048,512] f32
  const float* cb = (const float*)d_in[1];   // [4096,512]   f32
  float* out   = (float*)d_out;
  float* z_st  = out;                              // [N,512]
  float* z_bar = out + (size_t)N_ROWS * DIMS;      // [N,512]
  float* oidx  = out + (size_t)2 * N_ROWS * DIMS;  // [N] (as float)

  const size_t XE = (size_t)N_ROWS * DIMS;         // x elements
  const size_t CE = (size_t)KCODES * DIMS;         // codebook elements
  const size_t need = (2*XE + 2*CE) * sizeof(bf16_t) + KCODES * sizeof(float);

  if (ws_size >= need) {
    // fast path: pre-split bf16 planes in workspace, async-LDS GEMM
    bf16_t* xh = (bf16_t*)d_ws;
    bf16_t* xl = xh + XE;
    bf16_t* ch = xl + XE;
    bf16_t* cl = ch + CE;
    float* cnorm = (float*)(cl + CE);
    vq_split_kernel<<<(int)(XE/4 + 255)/256, 256, 0, stream>>>(x, xh, xl, (int)(XE/4));
    vq_split_kernel<<<(int)(CE/4 + 255)/256, 256, 0, stream>>>(cb, ch, cl, (int)(CE/4));
    vq_cnorm_kernel<<<KCODES/256, 256, 0, stream>>>(cb, cnorm);
    vq_argmin_fast_kernel<<<N_ROWS/64, 256, 0, stream>>>(xh, xl, ch, cl, cnorm, oidx);
  } else {
    // fallback: self-contained conversion in the GEMM kernel (needs 16KB ws)
    float* cnorm = (float*)d_ws;
    vq_cnorm_kernel<<<KCODES/256, 256, 0, stream>>>(cb, cnorm);
    vq_argmin_fallback_kernel<<<N_ROWS/64, 256, 0, stream>>>(x, cb, cnorm, oidx);
  }
  vq_gather_kernel<<<N_ROWS, 128, 0, stream>>>(cb, oidx, z_st, z_bar);
}
